// GCNClassifier_63359357551098
// MI455X (gfx1250) — matrix operand order repaired and checked by reading, not compiled, not run on hardware
//
#include <hip/hip_runtime.h>
#include <hip/hip_bf16.h>

// ---------------------------------------------------------------------------
// GCN classifier for MI455X (gfx1250, wave32).
// Dense GEMMs run on the WMMA pipe (v_wmma_f32_16x16x32_bf16, f32 accum),
// with all M dimensions padded to multiples of 16 so the WMMA kernel has a
// single non-divergent path (EXEC all-1s, as WMMA requires).
// Edge aggregation is fp32 scatter-add via native global_atomic_add_f32 —
// the 25.6MB feature buffers are resident in the 192MB L2, so atomics stay
// on-chip. Memory floor ~3.4GB of edge traffic => ~150us @ 23.3 TB/s.
// ---------------------------------------------------------------------------

#define BN_EPS 1e-5f

typedef __attribute__((ext_vector_type(16))) __bf16 v16bf;
typedef __attribute__((ext_vector_type(8)))  float  v8f;

// ---------------- utility ----------------
__global__ void k_fill(float* __restrict__ p, float v, int n) {
  int i = blockIdx.x * blockDim.x + threadIdx.x;
  if (i < n) p[i] = v;
}

// degree accumulation over edge targets (deg pre-initialized to 1.0 = self loop)
__global__ void k_deg(const int* __restrict__ dst, float* __restrict__ deg, int E) {
  int i = blockIdx.x * blockDim.x + threadIdx.x;
  if (i < E) unsafeAtomicAdd(&deg[dst[i]], 1.0f);
}

__global__ void k_rsqrt_inplace(float* __restrict__ p, int n) {
  int i = blockIdx.x * blockDim.x + threadIdx.x;
  if (i < n) p[i] = rsqrtf(p[i]);   // deg >= 1 always (self loop)
}

// ---------------- WMMA GEMM: C[M,N] = A[M,K] * B[K,N], fp32 in, bf16 WMMA ----
// blockDim = 128 (4 waves). Wave w computes tile (blockIdx.x, blockIdx.y*4+w).
// REQUIRES: M % 16 == 0, K % 32 == 0, N % 64 == 0 (callers pad).
__global__ void k_gemm_wmma(const float* __restrict__ A, const float* __restrict__ B,
                            float* __restrict__ C, int M, int K, int N) {
  const int lane = threadIdx.x & 31;
  const int wave = threadIdx.x >> 5;
  const int half = lane >> 4;       // 0: lanes 0-15, 1: lanes 16-31
  const int idx  = lane & 15;
  const int row0 = blockIdx.x * 16;
  const int col0 = (blockIdx.y * 4 + wave) * 16;
  const int r    = row0 + idx;

  v8f acc = {};
  for (int kb = 0; kb < K; kb += 32) {
    // A fragment: 16-bit A-matrix 16x32 layout (ISA 7.12.2):
    //  lane-half h holds rows r, K = kb+8h..kb+8h+7 and kb+16+8h..kb+16+8h+7.
    //  Two 8-float runs -> four 16B-aligned float4 loads (global_load_b128).
    const float* ap = A + r * K + kb + half * 8;
    const float4 a0 = *(const float4*)(ap);
    const float4 a1 = *(const float4*)(ap + 4);
    const float4 a2 = *(const float4*)(ap + 16);
    const float4 a3 = *(const float4*)(ap + 20);
    v16bf a;
    a[0]  = (__bf16)a0.x; a[1]  = (__bf16)a0.y; a[2]  = (__bf16)a0.z; a[3]  = (__bf16)a0.w;
    a[4]  = (__bf16)a1.x; a[5]  = (__bf16)a1.y; a[6]  = (__bf16)a1.z; a[7]  = (__bf16)a1.w;
    a[8]  = (__bf16)a2.x; a[9]  = (__bf16)a2.y; a[10] = (__bf16)a2.z; a[11] = (__bf16)a2.w;
    a[12] = (__bf16)a3.x; a[13] = (__bf16)a3.y; a[14] = (__bf16)a3.z; a[15] = (__bf16)a3.w;

    // B fragment: K rows striped across VGPRs, N across lanes (coalesced):
    //  lane-half h holds K=kb+8h..kb+8h+7 and kb+16+8h..kb+16+8h+7 (matches A).
    v16bf b;
#pragma unroll
    for (int e = 0; e < 16; ++e)
      b[e] = (__bf16)B[(kb + half * 8 + (e & 7) + ((e >> 3) << 4)) * N + col0 + idx];

    acc = __builtin_amdgcn_wmma_f32_16x16x32_bf16(
        /*neg_a=*/false, a, /*neg_b=*/false, b,
        /*c_mod=*/(short)0, acc, /*reuse_a=*/false, /*reuse_b=*/false);
  }
  // C/D layout: VGPR j holds (m = j + 8*half, n = idx)
#pragma unroll
  for (int j = 0; j < 8; ++j)
    C[(row0 + j + 8 * half) * N + col0 + idx] = acc[j];
}

// ---------------- aggregation ----------------
// init: self-loop contribution + bias:  out[i,ch] = xw[i,ch]*dinv[i]^2 + b[ch]
__global__ void k_agg_init(const float* __restrict__ xw, const float* __restrict__ dinv,
                           const float* __restrict__ bias, float* __restrict__ out, int total) {
  int i = blockIdx.x * blockDim.x + threadIdx.x;
  if (i < total) {
    int node = i >> 7;
    int ch   = i & 127;
    float di = dinv[node];
    out[i] = xw[i] * di * di + bias[ch];
  }
}

// edge scatter: one thread per (edge, 4-channel group). 128ch -> 32 groups.
__global__ void k_agg_edges(const int* __restrict__ src, const int* __restrict__ dst,
                            const float* __restrict__ dinv, const float* __restrict__ xw,
                            float* __restrict__ out, int E) {
  int tid = blockIdx.x * blockDim.x + threadIdx.x;
  if (tid >= E * 32) return;
  int e = tid >> 5;
  int g = tid & 31;
  int s = src[e], d = dst[e];
  float coef = dinv[s] * dinv[d];
  const float4 v = *(const float4*)(xw + s * 128 + g * 4);
  float* o = out + d * 128 + g * 4;
  unsafeAtomicAdd(o + 0, v.x * coef);
  unsafeAtomicAdd(o + 1, v.y * coef);
  unsafeAtomicAdd(o + 2, v.z * coef);
  unsafeAtomicAdd(o + 3, v.w * coef);
}

// ---------------- batchnorm over node dim (C = 128) ----------------
// Each 256-thread block covers 64 rows; LDS tree reduce -> few atomics.
__global__ void k_bn_reduce(const float* __restrict__ h, float* __restrict__ sums,
                            float* __restrict__ sqs, int Nn) {
  const int ROWS = 64;
  int ch   = threadIdx.x & 127;
  int rsub = threadIdx.x >> 7;             // 0 or 1
  int r0   = blockIdx.x * ROWS;
  float s = 0.f, q = 0.f;
  for (int i = rsub; i < ROWS; i += 2) {
    int r = r0 + i;
    if (r < Nn) {
      float v = h[r * 128 + ch];
      s += v; q += v * v;
    }
  }
  __shared__ float ss[256], qq[256];
  ss[threadIdx.x] = s; qq[threadIdx.x] = q;
  __syncthreads();
  if (threadIdx.x < 128) {
    unsafeAtomicAdd(&sums[ch], ss[threadIdx.x] + ss[threadIdx.x + 128]);
    unsafeAtomicAdd(&sqs[ch],  qq[threadIdx.x] + qq[threadIdx.x + 128]);
  }
}

__global__ void k_bn_apply_relu(float* __restrict__ h, const float* __restrict__ sums,
                                const float* __restrict__ sqs, const float* __restrict__ gamma,
                                const float* __restrict__ beta, int Nn, int total) {
  int i = blockIdx.x * blockDim.x + threadIdx.x;
  if (i < total) {
    int ch = i & 127;
    float invN = 1.0f / (float)Nn;
    float m = sums[ch] * invN;
    float v = sqs[ch] * invN - m * m;      // biased variance (jnp.var default)
    float y = (h[i] - m) * rsqrtf(v + BN_EPS) * gamma[ch] + beta[ch];
    h[i] = fmaxf(y, 0.0f);
  }
}

// ---------------- global mean pool ----------------
__global__ void k_pool_cnt(const int* __restrict__ batch, float* __restrict__ cnt, int Nn) {
  int i = blockIdx.x * blockDim.x + threadIdx.x;
  if (i < Nn) unsafeAtomicAdd(&cnt[batch[i]], 1.0f);
}

__global__ void k_pool_accum(const float* __restrict__ h, const int* __restrict__ batch,
                             float* __restrict__ pooled, int Nn) {
  int tid = blockIdx.x * blockDim.x + threadIdx.x;
  if (tid >= Nn * 32) return;
  int node = tid >> 5;
  int g    = tid & 31;
  int gr   = batch[node];
  const float4 v = *(const float4*)(h + node * 128 + g * 4);
  float* o = pooled + gr * 128 + g * 4;
  unsafeAtomicAdd(o + 0, v.x);
  unsafeAtomicAdd(o + 1, v.y);
  unsafeAtomicAdd(o + 2, v.z);
  unsafeAtomicAdd(o + 3, v.w);
}

__global__ void k_pool_div(float* __restrict__ pooled, const float* __restrict__ cnt, int total) {
  int i = blockIdx.x * blockDim.x + threadIdx.x;
  if (i < total) {
    int g = i >> 7;
    pooled[i] /= fmaxf(cnt[g], 1.0f);
  }
}

// ---------------- head BN+ReLU over 500 graphs, 256 channels ----------------
// b3 is omitted: a per-channel constant bias cancels exactly in batchnorm.
__global__ void k_bn_head(float* __restrict__ z, const float* __restrict__ gamma,
                          const float* __restrict__ beta, int G) {
  int ch = blockIdx.x * blockDim.x + threadIdx.x;
  if (ch >= 256) return;
  float s = 0.f, q = 0.f;
  for (int r = 0; r < G; ++r) {
    float v = z[r * 256 + ch];
    s += v; q += v * v;
  }
  float invG = 1.0f / (float)G;
  float m = s * invG;
  float var = q * invG - m * m;
  float sc = rsqrtf(var + BN_EPS) * gamma[ch];
  float bt = beta[ch];
  for (int r = 0; r < G; ++r) {
    float y = (z[r * 256 + ch] - m) * sc + bt;
    z[r * 256 + ch] = fmaxf(y, 0.0f);
  }
}

// ---------------- final tiny GEMM: out[G,10] = z[G,256] @ W4[256,10] + b4 ----
__global__ void k_final(const float* __restrict__ z, const float* __restrict__ W4,
                        const float* __restrict__ b4, float* __restrict__ out, int G) {
  int i = blockIdx.x * blockDim.x + threadIdx.x;
  if (i >= G * 10) return;
  int g = i / 10, c = i % 10;
  float acc = b4[c];
  for (int k = 0; k < 256; ++k) acc += z[g * 256 + k] * W4[k * 10 + c];
  out[i] = acc;
}

// ---------------------------------------------------------------------------
extern "C" void kernel_launch(void* const* d_in, const int* in_sizes, int n_in,
                              void* d_out, int out_size, void* d_ws, size_t ws_size,
                              hipStream_t stream) {
  // inputs (setup_inputs order)
  const float* x      = (const float*)d_in[0];    // [N,64]
  const int*   eidx   = (const int*)d_in[1];      // [2,E]
  const int*   batch  = (const int*)d_in[2];      // [N]
  const float* W1     = (const float*)d_in[3];    // [64,128]
  const float* b1     = (const float*)d_in[4];
  const float* gamma1 = (const float*)d_in[5];
  const float* beta1  = (const float*)d_in[6];
  const float* W2     = (const float*)d_in[7];    // [128,128]
  const float* b2     = (const float*)d_in[8];
  const float* gamma2 = (const float*)d_in[9];
  const float* beta2  = (const float*)d_in[10];
  const float* W3     = (const float*)d_in[11];   // [128,256]
  /* b3 = d_in[12] cancels under batchnorm */
  const float* gamma3 = (const float*)d_in[13];
  const float* beta3  = (const float*)d_in[14];
  const float* W4     = (const float*)d_in[15];   // [256,10]
  const float* b4     = (const float*)d_in[16];

  const int N  = in_sizes[0] / 64;    // 50000 (multiple of 16)
  const int E  = in_sizes[1] / 2;     // 1600000
  const int G  = 500;
  const int Gp = 512;                 // G padded to multiple of 16 for WMMA
  const int* src = eidx;
  const int* dst = eidx + E;

  // workspace layout (floats)
  float* ws     = (float*)d_ws;
  float* bufA   = ws;                        // N*128 : xw (GEMM output)
  float* bufB   = bufA + (size_t)N * 128;    // N*128 : aggregated / activated h
  float* dinv   = bufB + (size_t)N * 128;    // N
  float* bnsum  = dinv + N;                  // 128
  float* bnsq   = bnsum + 128;               // 128
  float* pooled = bnsq + 128;                // Gp*128 (rows 500..511 stay zero)
  float* cnt    = pooled + (size_t)Gp * 128; // G
  float* zbuf   = cnt + G;                   // Gp*256 (rows 500..511 garbage, unread)

  const int TPB = 256;
  const int elemsNC = N * 128;

  // --- degrees -> dinv (deg initialized to 1.0 for the self loop) ---
  k_fill<<<(N + TPB - 1) / TPB, TPB, 0, stream>>>(dinv, 1.0f, N);
  k_deg<<<(E + TPB - 1) / TPB, TPB, 0, stream>>>(dst, dinv, E);
  k_rsqrt_inplace<<<(N + TPB - 1) / TPB, TPB, 0, stream>>>(dinv, N);

  // ================= layer 1 =================
  // xw1 = x @ W1   [N,64]x[64,128]
  k_gemm_wmma<<<dim3(N / 16, 128 / 64), 128, 0, stream>>>(x, W1, bufA, N, 64, 128);
  k_agg_init<<<(elemsNC + TPB - 1) / TPB, TPB, 0, stream>>>(bufA, dinv, b1, bufB, elemsNC);
  k_agg_edges<<<(E * 32 + TPB - 1) / TPB, TPB, 0, stream>>>(src, dst, dinv, bufA, bufB, E);
  k_fill<<<1, 256, 0, stream>>>(bnsum, 0.0f, 256);  // zeros bnsum+bnsq (contiguous)
  k_bn_reduce<<<(N + 63) / 64, 256, 0, stream>>>(bufB, bnsum, bnsq, N);
  k_bn_apply_relu<<<(elemsNC + TPB - 1) / TPB, TPB, 0, stream>>>(bufB, bnsum, bnsq,
                                                                 gamma1, beta1, N, elemsNC);

  // ================= layer 2 =================
  // xw2 = h1 @ W2   [N,128]x[128,128]
  k_gemm_wmma<<<dim3(N / 16, 128 / 64), 128, 0, stream>>>(bufB, W2, bufA, N, 128, 128);
  k_agg_init<<<(elemsNC + TPB - 1) / TPB, TPB, 0, stream>>>(bufA, dinv, b2, bufB, elemsNC);
  k_agg_edges<<<(E * 32 + TPB - 1) / TPB, TPB, 0, stream>>>(src, dst, dinv, bufA, bufB, E);
  k_fill<<<1, 256, 0, stream>>>(bnsum, 0.0f, 256);
  k_bn_reduce<<<(N + 63) / 64, 256, 0, stream>>>(bufB, bnsum, bnsq, N);
  k_bn_apply_relu<<<(elemsNC + TPB - 1) / TPB, TPB, 0, stream>>>(bufB, bnsum, bnsq,
                                                                 gamma2, beta2, N, elemsNC);

  // ================= global mean pool (padded to 512 rows) =================
  k_fill<<<(Gp * 128 + G + TPB - 1) / TPB, TPB, 0, stream>>>(pooled, 0.0f, Gp * 128 + G);
  k_pool_cnt<<<(N + TPB - 1) / TPB, TPB, 0, stream>>>(batch, cnt, N);
  k_pool_accum<<<(N * 32 + TPB - 1) / TPB, TPB, 0, stream>>>(bufB, batch, pooled, N);
  k_pool_div<<<(G * 128 + TPB - 1) / TPB, TPB, 0, stream>>>(pooled, cnt, G * 128);

  // ================= MLP head =================
  // z = pooled @ W3   [512,128]x[128,256]  (b3 cancels in BN; pad rows are zero)
  k_gemm_wmma<<<dim3(Gp / 16, 256 / 64), 128, 0, stream>>>(pooled, W3, zbuf, Gp, 128, 256);
  k_bn_head<<<1, 256, 0, stream>>>(zbuf, gamma3, beta3, G);
  k_final<<<(G * 10 + TPB - 1) / TPB, TPB, 0, stream>>>(zbuf, W4, b4, (float*)d_out, G);
}
